// MDNLoss_29403346109042
// MI455X (gfx1250) — matrix operand order repaired and checked by reading, not compiled, not run on hardware
//
#include <hip/hip_runtime.h>
#include <hip/hip_bf16.h>

#define N_MEL   80
#define LDIM    256
#define TDIM    2048
#define BATCH   16
#define KTOT    160                 // 2*N_MEL folded GEMM depth
#define NEG_BIG (-1.0e15f)
#define LOG_2PI 1.8378770664093453f

typedef __bf16 v8bf  __attribute__((ext_vector_type(8)));
typedef __bf16 v16bf __attribute__((ext_vector_type(16)));
typedef float  v8f   __attribute__((ext_vector_type(8)));

__device__ __forceinline__ v16bf cat8(v8bf lo, v8bf hi) {
    return __builtin_shufflevector(lo, hi, 0,1,2,3,4,5,6,7,8,9,10,11,12,13,14,15);
}

// One workgroup per batch. 256 threads = 8 wave32. Wave w owns l-tiles 2w, 2w+1.
__global__ __launch_bounds__(256, 1)
void mdn_forward_kernel(const float* __restrict__ mu_sigma,
                        const float* __restrict__ melspec,
                        const int*   __restrict__ text_lengths,
                        const int*   __restrict__ mel_lengths,
                        float*       __restrict__ per_batch)
{
    __shared__ __align__(16) __bf16 stageA[32 * KTOT];     // 10 KB (A staging, one wave at a time)
    __shared__ __align__(16) __bf16 Bt[16 * KTOT];         //  5 KB (B^T for current t-tile)
    __shared__ __align__(16) float  lp_tile[16 * LDIM];    // 16 KB (lp[t_local][l])
    __shared__               float  abuf[2][LDIM];         //  2 KB (alpha ping-pong)

    const int b    = blockIdx.x;
    const int tid  = threadIdx.x;
    const int lane = tid & 31;
    const int wave = tid >> 5;
    const int hi   = (lane >= 16);

    // ---------------- Phase 1: per-l preprocessing + A fragments in registers ----
    // A row l (K=160): [ -0.5*inv_var[0..79] | mu*inv_var[0..79] ]
    v16bf afrag[2][5];
    float cterm = 0.0f;   // -0.5*sum(mu^2*inv_var) - log_coef, per thread (= per l)

    for (int r = 0; r < 8; ++r) {
        if (wave == r) {
            const float* row = mu_sigma + ((size_t)b * LDIM + tid) * KTOT;
            float s_mu2 = 0.0f, s_ls = 0.0f;
            for (int j = 0; j < N_MEL; ++j) {
                float m  = row[j];
                float ls = row[N_MEL + j];
                float mu = 1.0f / (1.0f + __expf(-m));
                float iv = __expf(-2.0f * ls);
                stageA[lane * KTOT + j]         = (__bf16)(-0.5f * iv);
                stageA[lane * KTOT + N_MEL + j] = (__bf16)(mu * iv);
                s_mu2 += mu * mu * iv;
                s_ls  += ls;
            }
            cterm = -0.5f * s_mu2 - (0.5f * (float)N_MEL * LOG_2PI + 0.5f * s_ls);
        }
        __syncthreads();
        if (wave == r) {
            // A(16x32 bf16) layout: lanes0-15: K=0..7 (elems0-7), K=16..23 (elems8-15)
            //                       lanes16-31: K=8..15,          K=24..31
            #pragma unroll
            for (int lt = 0; lt < 2; ++lt) {
                #pragma unroll
                for (int kk = 0; kk < 5; ++kk) {
                    int rloc = lt * 16 + (lane & 15);
                    int kb   = 32 * kk + (hi ? 8 : 0);
                    v8bf lo = *(const v8bf*)&stageA[rloc * KTOT + kb];
                    v8bf hh = *(const v8bf*)&stageA[rloc * KTOT + kb + 16];
                    afrag[lt][kk] = cat8(lo, hh);
                }
            }
        }
        __syncthreads();
    }

    // ---------------- Phase 2: fused GEMM (WMMA) + forward recursion ------------
    const int   tlen = text_lengths[b];
    const int   mlen = mel_lengths[b];
    const float* ms  = melspec + (size_t)b * N_MEL * TDIM;

    const int N      = lane & 15;          // local t column for WMMA B/D
    const int koffB  = hi ? 8 : 0;         // B(32x16): lanes0-15 K=0..7,16..23, lanes16-31 K=8..15,24..31
    const int lbase0 = wave * 32 + (hi ? 8 : 0);

    float a = NEG_BIG;                     // alpha for this thread's l

    for (int tile = 0; tile < TDIM / 16; ++tile) {
        const int t0 = tile * 16;

        // Build B^T: Bt[n][k] = (k<80) ? x[t0+n][k]^2 : x[t0+n][k-80]
        for (int i = tid; i < 16 * KTOT; i += 256) {
            int   n   = i / KTOT;
            int   k   = i - n * KTOT;
            int   mel = (k < N_MEL) ? k : (k - N_MEL);
            float v   = ms[(size_t)mel * TDIM + t0 + n];
            Bt[i] = (__bf16)((k < N_MEL) ? (v * v) : v);
        }
        __syncthreads();

        // lp tile = A(256x160) x B(160x16) via 2 l-tiles x 5 k-steps per wave
        v8f acc0 = {0,0,0,0,0,0,0,0};
        v8f acc1 = {0,0,0,0,0,0,0,0};
        #pragma unroll
        for (int kk = 0; kk < 5; ++kk) {
            v8bf blo = *(const v8bf*)&Bt[N * KTOT + 32 * kk + koffB];
            v8bf bhi = *(const v8bf*)&Bt[N * KTOT + 32 * kk + koffB + 16];
            v16bf bf = cat8(blo, bhi);
            acc0 = __builtin_amdgcn_wmma_f32_16x16x32_bf16(
                       false, afrag[0][kk], false, bf, (short)0, acc0, false, false);
            acc1 = __builtin_amdgcn_wmma_f32_16x16x32_bf16(
                       false, afrag[1][kk], false, bf, (short)0, acc1, false, false);
        }

        // D(16x16 f32): VGPR r -> lanes0-15: M=r, lanes16-31: M=8+r; N=lane&15
        #pragma unroll
        for (int r8 = 0; r8 < 8; ++r8) {
            lp_tile[N * LDIM + lbase0 + r8]      = acc0[r8];
            lp_tile[N * LDIM + lbase0 + 16 + r8] = acc1[r8];
        }
        __syncthreads();

        // 16 forward-recursion steps out of LDS (1 barrier/step, ping-pong alpha).
        // logaddexp via hardware transcendentals (v_exp_f32 + v_log_f32): this is
        // the 2048-step serial critical path, so the short sequence matters.
        for (int n2 = 0; n2 < 16; ++n2) {
            const int   t  = t0 + n2;
            const float lp = lp_tile[n2 * LDIM + tid] + cterm;
            if (t == 0) {
                a = (tid == 0) ? lp : NEG_BIG;
            } else {
                // shifted[0] = 0.0 exactly as reference (zeros concat)
                float s = (tid == 0) ? 0.0f : abuf[(t & 1) ^ 1][tid - 1];
                float m = fmaxf(a, s);
                float d = fabsf(a - s);
                a = m + __logf(1.0f + __expf(-d)) + lp;
            }
            abuf[t & 1][tid] = a;
            if (t == mlen - 1 && tid == tlen - 1) per_batch[b] = a;
            __syncthreads();
        }
    }
}

__global__ void mdn_finalize_kernel(const float* __restrict__ per_batch,
                                    float* __restrict__ out)
{
    if (threadIdx.x == 0) {
        float s = 0.0f;
        #pragma unroll
        for (int i = 0; i < BATCH; ++i) s += per_batch[i];
        out[0] = -s / (float)BATCH;
    }
}

extern "C" void kernel_launch(void* const* d_in, const int* in_sizes, int n_in,
                              void* d_out, int out_size, void* d_ws, size_t ws_size,
                              hipStream_t stream)
{
    const float* mu_sigma     = (const float*)d_in[0];
    const float* melspec      = (const float*)d_in[1];
    const int*   text_lengths = (const int*)d_in[2];
    const int*   mel_lengths  = (const int*)d_in[3];
    float*       per_batch    = (float*)d_ws;   // 16 floats of scratch

    mdn_forward_kernel<<<BATCH, 256, 0, stream>>>(mu_sigma, melspec,
                                                  text_lengths, mel_lengths,
                                                  per_batch);
    mdn_finalize_kernel<<<1, 32, 0, stream>>>(per_batch, (float*)d_out);
}
